// SwinTransformerBlock_14104672600653
// MI455X (gfx1250) — matrix operand order repaired and checked by reading, not compiled, not run on hardware
//
#include <hip/hip_runtime.h>
#include <hip/hip_bf16.h>

typedef __bf16 bf16_t;
typedef bf16_t v16bf __attribute__((ext_vector_type(16)));
typedef float  v8f   __attribute__((ext_vector_type(8)));
typedef unsigned short u16;
typedef unsigned int   u32;

#define BIMG   32
#define CCH    128
#define NHEAD  4
#define HDIM   32
#define NTOK   49          // 7*7 tokens per window
#define NWIN   64          // 8*8 windows per image
#define MTOK   100352      // BIMG*56*56 tokens (window-major order for attn path)
#define MT16   6272        // MTOK/16

// ---------- helpers ----------
__device__ __forceinline__ u16 f2bf(float f) {
    u32 u = __float_as_uint(f);
    u += 0x7fffu + ((u >> 16) & 1u);
    return (u16)(u >> 16);
}

union TileU { v16bf v; uint4 q[2]; };

// A-fragment (16x32 bf16): lane m = lane%16, half = lane/16.
// element i -> K = 16*(i/8) + 8*half + (i%8): two contiguous 16B chunks.
__device__ __forceinline__ v16bf load_tA(const u16* p, int lh) {
    TileU t;
    t.q[0] = *(const uint4*)(p + 8 * lh);
    t.q[1] = *(const uint4*)(p + 16 + 8 * lh);
    return t.v;
}
// B-fragment (32x16 bf16): lane n = lane%16, half = lane/16.
// element i -> K = 16*(i/8) + 8*half + (i%8): two contiguous 16B chunks.
__device__ __forceinline__ v16bf load_tB(const u16* p, int lh) {
    TileU t;
    t.q[0] = *(const uint4*)(p + 8 * lh);
    t.q[1] = *(const uint4*)(p + 16 + 8 * lh);
    return t.v;
}

__device__ __forceinline__ v8f wmma_bf(v16bf a, v16bf b, v8f c) {
    return __builtin_amdgcn_wmma_f32_16x16x32_bf16(false, a, false, b, (short)0, c, false, false);
}

// ---------- kernel 1: weight conversion f32 -> bf16 ----------
__global__ void k_cvtw(const float* __restrict__ qkvw, const float* __restrict__ projw,
                       const float* __restrict__ f1w, const float* __restrict__ f2w,
                       u16* __restrict__ wq, u16* __restrict__ wp,
                       u16* __restrict__ w1, u16* __restrict__ w2) {
    int i = blockIdx.x * 256 + threadIdx.x;   // grid covers 196608 exactly
    if (i < 49152)        wq[i]           = f2bf(qkvw[i]);
    else if (i < 65536)   wp[i - 49152]   = f2bf(projw[i - 49152]);
    else if (i < 131072)  w1[i - 65536]   = f2bf(f1w[i - 65536]);
    else                  w2[i - 131072]  = f2bf(f2w[i - 131072]);
}

// ---------- kernel 2/7: LayerNorm (wave per token), optional shift+window map ----------
__global__ void k_ln(const float* __restrict__ x, const float* __restrict__ g,
                     const float* __restrict__ b, u16* __restrict__ out, int swinmap) {
    int lane = threadIdx.x & 31;
    int wid  = threadIdx.x >> 5;
    int t = blockIdx.x * 4 + wid;             // destination token (window order if swinmap)
    size_t src;
    if (swinmap) {
        int bimg = t / (NWIN * NTOK);
        int r    = t % (NWIN * NTOK);
        int win = r / NTOK, tok = r % NTOK;
        int wh = win >> 3, wn = win & 7;
        int ih = tok / 7,  iw = tok % 7;
        int sh = (wh * 7 + ih + 3) % 56;
        int sw = (wn * 7 + iw + 3) % 56;
        src = ((size_t)bimg * 3136 + sh * 56 + sw) * CCH;
    } else {
        src = (size_t)t * CCH;
    }
    float4 xv = *(const float4*)(x + src + lane * 4);
    float s = xv.x + xv.y + xv.z + xv.w;
    for (int m = 16; m >= 1; m >>= 1) s += __shfl_xor(s, m, 32);
    float mean = s * (1.0f / 128.0f);
    float dx = xv.x - mean, dy = xv.y - mean, dz = xv.z - mean, dw = xv.w - mean;
    float ss = dx * dx + dy * dy + dz * dz + dw * dw;
    for (int m = 16; m >= 1; m >>= 1) ss += __shfl_xor(ss, m, 32);
    float rs = rsqrtf(ss * (1.0f / 128.0f) + 1e-5f);
    float4 gv = *(const float4*)(g + lane * 4);
    float4 bv = *(const float4*)(b + lane * 4);
    uint2 o;
    o.x = (u32)f2bf(dx * rs * gv.x + bv.x) | ((u32)f2bf(dy * rs * gv.y + bv.y) << 16);
    o.y = (u32)f2bf(dz * rs * gv.z + bv.z) | ((u32)f2bf(dw * rs * gv.w + bv.w) << 16);
    *(uint2*)(out + (size_t)t * CCH + lane * 4) = o;
}

// ---------- kernel 3: fused rel-pos-bias + shift-mask table (64,4,64,64) ----------
__device__ __forceinline__ int regid(int p) { return p < 49 ? 0 : (p < 53 ? 1 : 2); }

__global__ void k_bias(const float* __restrict__ rpb, float* __restrict__ comb) {
    int i = blockIdx.x * 256 + threadIdx.x;   // grid covers 64*4*64*64 exactly
    int col = i & 63, m = (i >> 6) & 63, h = (i >> 12) & 3, w = i >> 14;
    float v;
    if (m >= NTOK || col >= NTOK) {
        v = -1e30f;                            // pad columns die in softmax
    } else {
        int wh = w >> 3, wn = w & 7;
        int mr = m / 7, mc = m % 7, cr = col / 7, cc = col % 7;
        int idm = 3 * regid(wh * 7 + mr) + regid(wn * 7 + mc);
        int idc = 3 * regid(wh * 7 + cr) + regid(wn * 7 + cc);
        float maskv = (idm != idc) ? -100.0f : 0.0f;
        int ridx = (mr - cr + 6) * 13 + (mc - cc + 6);
        v = rpb[ridx * NHEAD + h] + maskv;
    }
    comb[i] = v;
}

// ---------- kernel 4: qkv GEMM  (M=100352, K=128, N=384), scatter to Q/K/V^T ----------
__global__ void __launch_bounds__(128) k_gemm_qkv(
        const u16* __restrict__ A, const u16* __restrict__ W, const float* __restrict__ bias,
        u16* __restrict__ Q, u16* __restrict__ K, u16* __restrict__ VT) {
    int lane = threadIdx.x & 31, wid = threadIdx.x >> 5;
    int gw = blockIdx.x * 4 + wid;
    int mt = gw % MT16;
    int ns = gw / MT16;                       // 0..5  (64 cols each)
    int lh = lane >> 4, ln = lane & 15;
    v8f acc[4] = {};
    const u16* arow = A + (size_t)(mt * 16 + ln) * 128;
#pragma unroll
    for (int ks = 0; ks < 4; ++ks) {
        v16bf a = load_tA(arow + ks * 32, lh);
#pragma unroll
        for (int nt = 0; nt < 4; ++nt) {
            int col = ns * 64 + nt * 16 + ln;
            v16bf bm = load_tB(W + (size_t)col * 128 + ks * 32, lh);
            acc[nt] = wmma_bf(a, bm, acc[nt]);
        }
    }
#pragma unroll
    for (int nt = 0; nt < 4; ++nt) {
        int col = ns * 64 + nt * 16 + ln;
        float bv = bias[col];
        int s = col >> 7, hcol = col & 127;
        int head = hcol >> 5, d = hcol & 31;
#pragma unroll
        for (int j = 0; j < 8; ++j) {
            int m = mt * 16 + lh * 8 + j;
            int win = m / NTOK, tok = m % NTOK;
            float v = acc[nt][j] + bv;
            size_t qk = ((size_t)(win * NHEAD + head) * 64 + tok) * HDIM + d;
            if (s == 0)      Q[qk] = f2bf(v * 0.17677669529663687f);   // hd^-0.5
            else if (s == 1) K[qk] = f2bf(v);
            else             VT[((size_t)(win * NHEAD + head) * HDIM + d) * 64 + tok] = f2bf(v);
        }
    }
}

// ---------- kernel 5: windowed attention, one wave per (window, head) ----------
__global__ void __launch_bounds__(128) k_attn(
        const u16* __restrict__ Q, const u16* __restrict__ K, const u16* __restrict__ VT,
        const float* __restrict__ comb, u16* __restrict__ AOUT) {
    __shared__ u16 smem[4 * 64 * 80];         // P staging, padded stride 80
    int lane = threadIdx.x & 31, wid = threadIdx.x >> 5;
    int gw = blockIdx.x * 4 + wid;            // (win, head)
    int win = gw >> 2, head = gw & 3;
    int wloc = win & (NWIN - 1);
    int lh = lane >> 4, ln = lane & 15;
    u16* sp = smem + wid * 64 * 80;

    const u16* qb = Q  + (size_t)gw * 64 * HDIM;
    const u16* kb = K  + (size_t)gw * 64 * HDIM;
    const u16* vb = VT + (size_t)gw * HDIM * 64;

    v16bf qa[4], kbt[4];
#pragma unroll
    for (int i = 0; i < 4; ++i) qa[i]  = load_tA(qb + (size_t)(i * 16 + ln) * HDIM, lh);
#pragma unroll
    for (int i = 0; i < 4; ++i) kbt[i] = load_tB(kb + (size_t)(i * 16 + ln) * HDIM, lh);

    // S = q k^T  (64x64, K-dim = 32 -> single WMMA step)
    v8f S[4][4];
#pragma unroll
    for (int ms = 0; ms < 4; ++ms)
#pragma unroll
        for (int nsb = 0; nsb < 4; ++nsb) {
            v8f z = {};
            S[ms][nsb] = wmma_bf(qa[ms], kbt[nsb], z);
        }

    // bias/mask add + row softmax (rows live in a fixed half-wave; reduce over 16 lanes)
    const float* cb = comb + (size_t)(wloc * NHEAD + head) * 64 * 64;
#pragma unroll
    for (int ms = 0; ms < 4; ++ms) {
#pragma unroll
        for (int j = 0; j < 8; ++j) {
            int row = ms * 16 + lh * 8 + j;
            float vals[4];
#pragma unroll
            for (int nsb = 0; nsb < 4; ++nsb)
                vals[nsb] = S[ms][nsb][j] + cb[row * 64 + nsb * 16 + ln];
            float mx = fmaxf(fmaxf(vals[0], vals[1]), fmaxf(vals[2], vals[3]));
            for (int msk = 8; msk >= 1; msk >>= 1) mx = fmaxf(mx, __shfl_xor(mx, msk, 32));
            float sum = 0.0f;
#pragma unroll
            for (int nsb = 0; nsb < 4; ++nsb) { vals[nsb] = __expf(vals[nsb] - mx); sum += vals[nsb]; }
            for (int msk = 8; msk >= 1; msk >>= 1) sum += __shfl_xor(sum, msk, 32);
            float inv = 1.0f / sum;
#pragma unroll
            for (int nsb = 0; nsb < 4; ++nsb)
                sp[row * 80 + nsb * 16 + ln] = f2bf(vals[nsb] * inv);
        }
    }

    // O = P v   (K-dim 64 -> two WMMA steps); P re-read from LDS in A layout
    v8f O[4][2] = {};
#pragma unroll
    for (int ks = 0; ks < 2; ++ks) {
        v16bf vbt[2];
#pragma unroll
        for (int nt = 0; nt < 2; ++nt)
            vbt[nt] = load_tB(vb + (size_t)(nt * 16 + ln) * 64 + ks * 32, lh);
#pragma unroll
        for (int ms = 0; ms < 4; ++ms) {
            v16bf pa = load_tA(sp + (ms * 16 + ln) * 80 + ks * 32, lh);
#pragma unroll
            for (int nt = 0; nt < 2; ++nt)
                O[ms][nt] = wmma_bf(pa, vbt[nt], O[ms][nt]);
        }
    }

#pragma unroll
    for (int ms = 0; ms < 4; ++ms)
#pragma unroll
        for (int j = 0; j < 8; ++j) {
            int m = ms * 16 + lh * 8 + j;
            if (m < NTOK) {
#pragma unroll
                for (int nt = 0; nt < 2; ++nt) {
                    int d = nt * 16 + ln;
                    AOUT[((size_t)win * NTOK + m) * CCH + head * HDIM + d] = f2bf(O[ms][nt][j]);
                }
            }
        }
}

// ---------- kernel 6: proj GEMM + window-reverse + unshift + residual -> x2 (f32) ----------
__global__ void __launch_bounds__(128) k_gemm_proj(
        const u16* __restrict__ A, const u16* __restrict__ W, const float* __restrict__ bias,
        const float* __restrict__ xin, float* __restrict__ x2) {
    int lane = threadIdx.x & 31, wid = threadIdx.x >> 5;
    int gw = blockIdx.x * 4 + wid;
    int mt = gw % MT16;
    int ns = gw / MT16;                       // 0..1
    int lh = lane >> 4, ln = lane & 15;
    v8f acc[4] = {};
    const u16* arow = A + (size_t)(mt * 16 + ln) * 128;
#pragma unroll
    for (int ks = 0; ks < 4; ++ks) {
        v16bf a = load_tA(arow + ks * 32, lh);
#pragma unroll
        for (int nt = 0; nt < 4; ++nt) {
            int col = ns * 64 + nt * 16 + ln;
            v16bf bm = load_tB(W + (size_t)col * 128 + ks * 32, lh);
            acc[nt] = wmma_bf(a, bm, acc[nt]);
        }
    }
#pragma unroll
    for (int nt = 0; nt < 4; ++nt) {
        int col = ns * 64 + nt * 16 + ln;
        float bv = bias[col];
#pragma unroll
        for (int j = 0; j < 8; ++j) {
            int m = mt * 16 + lh * 8 + j;
            int bimg = m / 3136, r = m % 3136;
            int winp = r / NTOK, tok = r % NTOK;
            int wh = winp >> 3, wn = winp & 7;
            int ih = tok / 7,   iw = tok % 7;
            int sh = (wh * 7 + ih + 3) % 56;
            int sw = (wn * 7 + iw + 3) % 56;
            size_t dst = ((size_t)bimg * 3136 + sh * 56 + sw) * CCH + col;
            x2[dst] = xin[dst] + acc[nt][j] + bv;
        }
    }
}

// ---------- kernel 8: fc1 GEMM + exact GELU -> bf16 (M,512) ----------
__global__ void __launch_bounds__(128) k_gemm_fc1(
        const u16* __restrict__ A, const u16* __restrict__ W, const float* __restrict__ bias,
        u16* __restrict__ out) {
    int lane = threadIdx.x & 31, wid = threadIdx.x >> 5;
    int gw = blockIdx.x * 4 + wid;
    int mt = gw % MT16;
    int ns = gw / MT16;                       // 0..7
    int lh = lane >> 4, ln = lane & 15;
    v8f acc[4] = {};
    const u16* arow = A + (size_t)(mt * 16 + ln) * 128;
#pragma unroll
    for (int ks = 0; ks < 4; ++ks) {
        v16bf a = load_tA(arow + ks * 32, lh);
#pragma unroll
        for (int nt = 0; nt < 4; ++nt) {
            int col = ns * 64 + nt * 16 + ln;
            v16bf bm = load_tB(W + (size_t)col * 128 + ks * 32, lh);
            acc[nt] = wmma_bf(a, bm, acc[nt]);
        }
    }
#pragma unroll
    for (int nt = 0; nt < 4; ++nt) {
        int col = ns * 64 + nt * 16 + ln;
        float bv = bias[col];
#pragma unroll
        for (int j = 0; j < 8; ++j) {
            int m = mt * 16 + lh * 8 + j;
            float v = acc[nt][j] + bv;
            v = 0.5f * v * (1.0f + erff(v * 0.70710678118654752f));
            out[(size_t)m * 512 + col] = f2bf(v);
        }
    }
}

// ---------- kernel 9: fc2 GEMM (K=512) + bias + residual -> d_out (f32) ----------
__global__ void __launch_bounds__(128) k_gemm_fc2(
        const u16* __restrict__ A, const u16* __restrict__ W, const float* __restrict__ bias,
        const float* __restrict__ x2, float* __restrict__ out) {
    int lane = threadIdx.x & 31, wid = threadIdx.x >> 5;
    int gw = blockIdx.x * 4 + wid;
    int mt = gw % MT16;
    int ns = gw / MT16;                       // 0..1
    int lh = lane >> 4, ln = lane & 15;
    v8f acc[4] = {};
    const u16* arow = A + (size_t)(mt * 16 + ln) * 512;
#pragma unroll
    for (int ks = 0; ks < 16; ++ks) {
        v16bf a = load_tA(arow + ks * 32, lh);
#pragma unroll
        for (int nt = 0; nt < 4; ++nt) {
            int col = ns * 64 + nt * 16 + ln;
            v16bf bm = load_tB(W + (size_t)col * 512 + ks * 32, lh);
            acc[nt] = wmma_bf(a, bm, acc[nt]);
        }
    }
#pragma unroll
    for (int nt = 0; nt < 4; ++nt) {
        int col = ns * 64 + nt * 16 + ln;
        float bv = bias[col];
#pragma unroll
        for (int j = 0; j < 8; ++j) {
            int m = mt * 16 + lh * 8 + j;
            size_t idx = (size_t)m * CCH + col;
            out[idx] = x2[idx] + acc[nt][j] + bv;
        }
    }
}

// ---------- launcher ----------
extern "C" void kernel_launch(void* const* d_in, const int* in_sizes, int n_in,
                              void* d_out, int out_size, void* d_ws, size_t ws_size,
                              hipStream_t stream) {
    (void)in_sizes; (void)n_in; (void)out_size; (void)ws_size;
    const float* x      = (const float*)d_in[0];
    const float* n1g    = (const float*)d_in[1];
    const float* n1b    = (const float*)d_in[2];
    const float* qkv_w  = (const float*)d_in[3];
    const float* qkv_b  = (const float*)d_in[4];
    const float* rpb    = (const float*)d_in[5];
    const float* proj_w = (const float*)d_in[6];
    const float* proj_b = (const float*)d_in[7];
    const float* n2g    = (const float*)d_in[8];
    const float* n2b    = (const float*)d_in[9];
    const float* fc1_w  = (const float*)d_in[10];
    const float* fc1_b  = (const float*)d_in[11];
    const float* fc2_w  = (const float*)d_in[12];
    const float* fc2_b  = (const float*)d_in[13];
    float* out = (float*)d_out;

    char* ws = (char*)d_ws;
    size_t o_wq   = 0;                          // qkv weight bf16   (98304 B)
    size_t o_wp   = o_wq   + 98304;             // proj weight bf16  (32768 B)
    size_t o_w1   = o_wp   + 32768;             // fc1 weight bf16   (131072 B)
    size_t o_w2   = o_w1   + 131072;            // fc2 weight bf16   (131072 B)
    size_t o_comb = o_w2   + 131072;            // bias+mask f32     (4194304 B)
    size_t o_hwin = o_comb + 4194304;           // ln1 out / attn out bf16 (25690112 B)
    size_t o_q    = o_hwin + 25690112;          // Q padded / h2 bf16 (33554432 B)
    size_t o_k    = o_q    + 33554432;          // K padded
    size_t o_vt   = o_k    + 33554432;          // V^T padded
    size_t o_x2   = o_vt   + 33554432;          // x2 f32            (51380224 B)
    size_t o_f1   = o_x2   + 51380224;          // fc1+gelu bf16     (102760448 B)

    u16*   WQ   = (u16*)(ws + o_wq);
    u16*   WP   = (u16*)(ws + o_wp);
    u16*   W1   = (u16*)(ws + o_w1);
    u16*   W2   = (u16*)(ws + o_w2);
    float* COMB = (float*)(ws + o_comb);
    u16*   HWIN = (u16*)(ws + o_hwin);          // reused as attention output
    u16*   Qb   = (u16*)(ws + o_q);             // reused as h2 after attention
    u16*   Kb   = (u16*)(ws + o_k);
    u16*   VTb  = (u16*)(ws + o_vt);
    float* X2   = (float*)(ws + o_x2);
    u16*   F1   = (u16*)(ws + o_f1);

    k_cvtw<<<768, 256, 0, stream>>>(qkv_w, proj_w, fc1_w, fc2_w, WQ, WP, W1, W2);
    k_ln<<<25088, 128, 0, stream>>>(x, n1g, n1b, HWIN, 1);
    k_bias<<<4096, 256, 0, stream>>>(rpb, COMB);
    k_gemm_qkv<<<9408, 128, 0, stream>>>(HWIN, WQ, qkv_b, Qb, Kb, VTb);
    k_attn<<<2048, 128, 0, stream>>>(Qb, Kb, VTb, COMB, HWIN);
    k_gemm_proj<<<3136, 128, 0, stream>>>(HWIN, WP, proj_b, x, X2);
    k_ln<<<25088, 128, 0, stream>>>(X2, n2g, n2b, Qb, 0);
    k_gemm_fc1<<<12544, 128, 0, stream>>>(Qb, W1, fc1_b, F1);
    k_gemm_fc2<<<3136, 128, 0, stream>>>(F1, W2, fc2_b, X2, out);
}
